// FrameTransformer_74878459838477
// MI455X (gfx1250) — matrix operand order repaired and checked by reading, not compiled, not run on hardware
//
#include <hip/hip_runtime.h>

// Problem constants (match reference)
#define BB    2
#define CCH   8
#define BINSN 1024
#define WLEN  512
#define NHEAD 16
#define HDIM  64
#define ROTD  32
#define GMLP  4096

typedef __attribute__((ext_vector_type(16))) __bf16 v16bf;
typedef __attribute__((ext_vector_type(8)))  __bf16 v8bf;
typedef __attribute__((ext_vector_type(8)))  float  v8f;
typedef __attribute__((ext_vector_type(4)))  int    v4i;

// ---------------------------------------------------------------------------
// CDNA5 async copy to LDS (guarded so compile never breaks)
// ---------------------------------------------------------------------------
#if defined(__AMDGCN__) && __has_builtin(__builtin_amdgcn_global_load_async_to_lds_b128) && __has_builtin(__builtin_amdgcn_s_wait_asynccnt)
#define USE_ASYNC_CP 1
#endif

__device__ __forceinline__ void async_cp_b128(const void* g, void* l) {
#ifdef USE_ASYNC_CP
  __builtin_amdgcn_global_load_async_to_lds_b128(
      (__attribute__((address_space(1))) v4i*)(g),
      (__attribute__((address_space(3))) v4i*)(l), 0, 0);
#else
  *(v8bf*)l = *(const v8bf*)g;
#endif
}
__device__ __forceinline__ void async_cp_wait() {
#ifdef USE_ASYNC_CP
  __builtin_amdgcn_s_wait_asynccnt(0);
#endif
}

// ---------------------------------------------------------------------------
// WMMA helpers (bf16 in, f32 accumulate), fragment loads in contiguous k order (k = 16*(lane/16) + s)
// ---------------------------------------------------------------------------
__device__ __forceinline__ v8f wmma_bf16(v16bf a, v16bf b, v8f c) {
  return __builtin_amdgcn_wmma_f32_16x16x32_bf16(false, a, false, b, (short)0, c,
                                                 false, false);
}

union FragU { v16bf v; v8bf h[2]; };

// A fragment (16x32): base[m][k] row-major with given stride.
// lane<16: K = k0..k0+15 ; lane>=16: K = k0+16..k0+31
__device__ __forceinline__ v16bf frag_a(const __bf16* base, int stride,
                                        int m_base, int k0, int lane) {
  const __bf16* r = base + (size_t)(m_base + (lane & 15)) * stride + k0 +
                    (((lane >> 4) & 1) << 4);
  FragU u;
  u.h[0] = *(const v8bf*)(r);
  u.h[1] = *(const v8bf*)(r + 8);
  return u.v;
}

// B fragment (32x16): base stored transposed [n][k] (k contiguous), stride = k-pitch.
// lane<16: K = k0..k0+15 ; lane>=16: K = k0+16..k0+31 ; column = n_base + (lane&15)
__device__ __forceinline__ v16bf frag_b(const __bf16* base, int stride,
                                        int n_base, int k0, int lane) {
  const __bf16* r = base + (size_t)(n_base + (lane & 15)) * stride + k0 +
                    (((lane >> 4) & 1) << 4);
  FragU u;
  u.h[0] = *(const v8bf*)(r);
  u.h[1] = *(const v8bf*)(r + 8);
  return u.v;
}

// ---------------------------------------------------------------------------
// Kernel: f32 -> bf16 conversion (weights)
// ---------------------------------------------------------------------------
__global__ __launch_bounds__(256) void cvt_bf16_kernel(const float* __restrict__ in,
                                                       __bf16* __restrict__ out,
                                                       long n) {
  long i = (long)blockIdx.x * 256 + threadIdx.x;
  if (i < n) out[i] = (__bf16)in[i];
}

// ---------------------------------------------------------------------------
// Kernel: frame norm over BINS axis for each (b,c,w) column -> bf16
// ---------------------------------------------------------------------------
__global__ __launch_bounds__(256) void frame_norm_kernel(
    const float* __restrict__ x, const float* __restrict__ gw,
    const float* __restrict__ gb, __bf16* __restrict__ out) {
  __shared__ float s1[256];
  __shared__ float s2[256];
  const int col = blockIdx.x;          // 0 .. B*C*W-1
  const int bc  = col >> 9;            // / WLEN
  const int wi  = col & (WLEN - 1);
  const float* xp = x + ((size_t)bc << 19) + wi;  // bc*BINS*W
  const int t = threadIdx.x;
  float v[4], sum = 0.f, ss = 0.f;
#pragma unroll
  for (int r = 0; r < 4; ++r) {
    int f = t + r * 256;
    v[r] = xp[(size_t)f * WLEN];
    sum += v[r];
    ss  += v[r] * v[r];
  }
  s1[t] = sum; s2[t] = ss;
  __syncthreads();
  for (int o = 128; o > 0; o >>= 1) {
    if (t < o) { s1[t] += s1[t + o]; s2[t] += s2[t + o]; }
    __syncthreads();
  }
  const float mean = s1[0] * (1.0f / BINSN);
  const float var  = s2[0] * (1.0f / BINSN) - mean * mean;
  const float rs   = rsqrtf(var + 1e-5f);
#pragma unroll
  for (int r = 0; r < 4; ++r) {
    int f = t + r * 256;
    out[((size_t)bc << 19) + (size_t)f * WLEN + wi] =
        (__bf16)((v[r] - mean) * rs * gw[f] + gb[f]);
  }
}

// ---------------------------------------------------------------------------
// Kernel: per-channel GEMM  Out[bc](G x N) = W[c](G x F) * In[bc](F x N)
//   A (weights) streamed via async-to-LDS b128; B transposed into LDS
//   128x128 block tile, K-step 32, 8 waves, 2x4 16x16 WMMA tiles per wave
//   mode: 0 = bf16 out, 1 = f32 out, 2 = exact-GELU -> bf16, 3 = resid + f32
// ---------------------------------------------------------------------------
__global__ __launch_bounds__(256) void gemm_bf16_kernel(
    const __bf16* __restrict__ Aw, const __bf16* __restrict__ Bin,
    const float* __restrict__ resid, void* __restrict__ outp, int G, int F,
    int N, int mode) {
  __shared__ __bf16 As[2][128 * 32];   // [m][k]
  __shared__ __bf16 Bs[2][128 * 32];   // [n][k]  (transposed)
  const int bc = blockIdx.z;
  const int ch = bc & (CCH - 1);
  const __bf16* A0 = Aw + (size_t)ch * G * F;
  const __bf16* B0 = Bin + (size_t)bc * F * N;
  const int g0 = blockIdx.x * 128;
  const int n0 = blockIdx.y * 128;
  const int t = threadIdx.x, lane = t & 31, wv = t >> 5;
  const int wm = wv >> 1, wn = wv & 1;

  v8f acc[2][4];
#pragma unroll
  for (int a = 0; a < 2; ++a)
#pragma unroll
    for (int b = 0; b < 4; ++b)
#pragma unroll
      for (int e = 0; e < 8; ++e) acc[a][b][e] = 0.0f;

  auto load_tiles = [&](int buf, int k0) {
    // A tile: 128x32 bf16 = 512 x 16B chunks, async direct-to-LDS
#pragma unroll
    for (int j = 0; j < 2; ++j) {
      int idx = (t << 1) + j;
      int m = idx >> 2, c4 = idx & 3;  // 4 x b128 per row
      async_cp_b128(A0 + (size_t)(g0 + m) * F + k0 + c4 * 8,
                    &As[buf][m * 32 + c4 * 8]);
    }
    // B tile: load (k,n) row-major chunks, store transposed [n][k]
#pragma unroll
    for (int j = 0; j < 2; ++j) {
      int idx = (t << 1) + j;
      int k = idx >> 4, nch = idx & 15;
      v8bf d = *(const v8bf*)(B0 + (size_t)(k0 + k) * N + n0 + nch * 8);
#pragma unroll
      for (int e = 0; e < 8; ++e) Bs[buf][(nch * 8 + e) * 32 + k] = d[e];
    }
  };

  load_tiles(0, 0);
  async_cp_wait();
  __syncthreads();

  const int steps = F >> 5;
  for (int s = 0; s < steps; ++s) {
    const int cur = s & 1;
    if (s + 1 < steps) load_tiles(cur ^ 1, (s + 1) * 32);
    if (s + 2 < steps)
      __builtin_prefetch(A0 + (size_t)(g0 + (t >> 1)) * F + (size_t)(s + 2) * 32,
                         0, 0);
    v16bf af[2], bf[4];
#pragma unroll
    for (int tm = 0; tm < 2; ++tm)
      af[tm] = frag_a(&As[cur][0], 32, wm * 32 + tm * 16, 0, lane);
#pragma unroll
    for (int tn = 0; tn < 4; ++tn)
      bf[tn] = frag_b(&Bs[cur][0], 32, wn * 64 + tn * 16, 0, lane);
#pragma unroll
    for (int tm = 0; tm < 2; ++tm)
#pragma unroll
      for (int tn = 0; tn < 4; ++tn)
        acc[tm][tn] = wmma_bf16(af[tm], bf[tn], acc[tm][tn]);
    async_cp_wait();
    __syncthreads();
  }

#pragma unroll
  for (int tm = 0; tm < 2; ++tm)
#pragma unroll
    for (int tn = 0; tn < 4; ++tn)
#pragma unroll
      for (int i = 0; i < 8; ++i) {
        const int g = g0 + wm * 32 + tm * 16 + i + ((lane >> 4) << 3);
        const int n = n0 + wn * 64 + tn * 16 + (lane & 15);
        const size_t o = ((size_t)bc * G + g) * N + n;
        const float v = acc[tm][tn][i];
        if (mode == 0)
          ((__bf16*)outp)[o] = (__bf16)v;
        else if (mode == 1)
          ((float*)outp)[o] = v;
        else if (mode == 2)
          ((__bf16*)outp)[o] =
              (__bf16)(0.5f * v * (1.0f + erff(v * 0.70710678118f)));
        else
          ((float*)outp)[o] = resid[o] + v;
      }
}

// ---------------------------------------------------------------------------
// Kernel: flash attention per (query-block, head, bc). q == k == rotate(y).
//   y layout: (bc, g=h*64+d, w) f32.  Output bf16 in same (bc, g, w) layout.
// ---------------------------------------------------------------------------
__global__ __launch_bounds__(256) void attn_kernel(const float* __restrict__ y,
                                                   const float* __restrict__ freqs,
                                                   __bf16* __restrict__ out) {
  extern __shared__ char smem_raw[];
  __bf16* Qs = (__bf16*)smem_raw;   // [128][64]  query rows (rotated)
  __bf16* Ks = Qs + 128 * 64;       // [128][64]  key rows   (rotated)
  __bf16* Vs = Ks + 128 * 64;       // [64][128]  V^T: [d][u] (natural y layout)
  __bf16* Ps = Vs + 64 * 128;       // [8 waves][16][128] softmax P staging

  const int qb = blockIdx.x, h = blockIdx.y, bc = blockIdx.z;
  const float* Y = y + ((size_t)bc * BINSN + h * HDIM) * WLEN;  // Y[d][w]
  const int t = threadIdx.x, lane = t & 31, wv = t >> 5, hi = lane >> 4;
  const int wq0 = qb * 128;

  float fr[16];
#pragma unroll
  for (int p = 0; p < 16; ++p) fr[p] = freqs[p];

  // Build rotated Q tile
  for (int i = t; i < 128 * 64; i += 256) {
    const int wl = i >> 6, d = i & 63;
    const int w = wq0 + wl;
    float v = Y[(size_t)d * WLEN + w];
    if (d < ROTD) {
      const float a = (float)w * fr[d >> 1];
      const float c = cosf(a), s = sinf(a);
      const float o = Y[(size_t)(d ^ 1) * WLEN + w];
      v = (d & 1) ? (v * c + o * s) : (v * c - o * s);
    }
    Qs[wl * 64 + d] = (__bf16)v;
  }
  __syncthreads();

  v8f O[4];
  float mi[8], li[8];
#pragma unroll
  for (int n = 0; n < 4; ++n)
#pragma unroll
    for (int e = 0; e < 8; ++e) O[n][e] = 0.0f;
#pragma unroll
  for (int i = 0; i < 8; ++i) { mi[i] = -3.0e38f; li[i] = 0.0f; }

  for (int kb = 0; kb < 4; ++kb) {
    __syncthreads();  // previous Ks/Vs consumers done
    const int u0 = kb * 128;
    for (int i = t; i < 128 * 64; i += 256) {
      const int ul = i >> 6, d = i & 63;
      const int u = u0 + ul;
      float v = Y[(size_t)d * WLEN + u];
      Vs[d * 128 + ul] = (__bf16)v;  // V^T
      if (d < ROTD) {
        const float a = (float)u * fr[d >> 1];
        const float c = cosf(a), s = sinf(a);
        const float o = Y[(size_t)(d ^ 1) * WLEN + u];
        v = (d & 1) ? (v * c + o * s) : (v * c - o * s);
      }
      Ks[ul * 64 + d] = (__bf16)v;
    }
    __syncthreads();

    // S = (Q Kt) / sqrt(BINS) : 8 u-tiles x 2 k-steps of WMMA
    v8f S[8];
#pragma unroll
    for (int j = 0; j < 8; ++j)
#pragma unroll
      for (int e = 0; e < 8; ++e) S[j][e] = 0.0f;
#pragma unroll
    for (int j = 0; j < 8; ++j)
#pragma unroll
      for (int ks = 0; ks < 2; ++ks)
        S[j] = wmma_bf16(frag_a(Qs, 64, wv * 16, ks * 32, lane),
                         frag_b(Ks, 64, j * 16, ks * 32, lane), S[j]);

    // online softmax over the 128 keys of this block
    float corr[8];
#pragma unroll
    for (int i = 0; i < 8; ++i) {
      float mx = S[0][i] * 0.03125f;
#pragma unroll
      for (int j = 1; j < 8; ++j) mx = fmaxf(mx, S[j][i] * 0.03125f);
#pragma unroll
      for (int msk = 1; msk < 16; msk <<= 1)
        mx = fmaxf(mx, __shfl_xor(mx, msk, 32));
      const float mnew = fmaxf(mi[i], mx);
      corr[i] = expf(mi[i] - mnew);
      float sum = 0.0f;
#pragma unroll
      for (int j = 0; j < 8; ++j) {
        const float p = expf(S[j][i] * 0.03125f - mnew);
        S[j][i] = p;
        sum += p;
      }
#pragma unroll
      for (int msk = 1; msk < 16; msk <<= 1) sum += __shfl_xor(sum, msk, 32);
      li[i] = li[i] * corr[i] + sum;
      mi[i] = mnew;
    }

    // stage P (D-frag layout) -> LDS -> A-frag layout
    __bf16* Pw = Ps + wv * 16 * 128;
#pragma unroll
    for (int j = 0; j < 8; ++j)
#pragma unroll
      for (int i = 0; i < 8; ++i)
        Pw[(i + hi * 8) * 128 + j * 16 + (lane & 15)] = (__bf16)S[j][i];

#pragma unroll
    for (int n = 0; n < 4; ++n)
#pragma unroll
      for (int i = 0; i < 8; ++i) O[n][i] *= corr[i];

    // O += P (16x128) * V (128x64)
#pragma unroll
    for (int n = 0; n < 4; ++n)
#pragma unroll
      for (int ks = 0; ks < 4; ++ks)
        O[n] = wmma_bf16(frag_a(Pw, 128, 0, ks * 32, lane),
                         frag_b(Vs, 128, n * 16, ks * 32, lane), O[n]);
  }

  // normalize and write back in (bc, g, w) layout
#pragma unroll
  for (int n = 0; n < 4; ++n)
#pragma unroll
    for (int i = 0; i < 8; ++i) {
      const int d = n * 16 + (lane & 15);
      const int w = wq0 + wv * 16 + i + hi * 8;
      out[((size_t)bc * BINSN + h * HDIM + d) * WLEN + w] =
          (__bf16)(O[n][i] / li[i]);
    }
}

// ---------------------------------------------------------------------------
// Host orchestration
// ---------------------------------------------------------------------------
extern "C" void kernel_launch(void* const* d_in, const int* in_sizes, int n_in,
                              void* d_out, int out_size, void* d_ws,
                              size_t ws_size, hipStream_t stream) {
  (void)in_sizes; (void)n_in; (void)out_size; (void)ws_size;
  const float* x        = (const float*)d_in[0];
  const float* norm1_w  = (const float*)d_in[1];
  const float* norm1_b  = (const float*)d_in[2];
  const float* rotf     = (const float*)d_in[3];
  const float* q_w      = (const float*)d_in[4];
  const float* out_w    = (const float*)d_in[5];
  const float* norm2_w  = (const float*)d_in[6];
  const float* norm2_b  = (const float*)d_in[7];
  const float* lin1_w   = (const float*)d_in[8];
  const float* lin2_w   = (const float*)d_in[9];
  float* outp = (float*)d_out;

  const long NQ = (long)CCH * BINSN * BINSN;     // 8.39M  (q_w / out_w)
  const long NL = (long)CCH * GMLP * BINSN;      // 33.6M  (lin1 / lin2)
  const long NX = (long)BB * CCH * BINSN * WLEN; // 8.39M  activations
  const long NH = (long)BB * CCH * GMLP * WLEN;  // 33.6M  MLP hidden

  char* p = (char*)d_ws;
  __bf16* wq = (__bf16*)p; p += NQ * 2;
  __bf16* wo = (__bf16*)p; p += NQ * 2;
  __bf16* w1 = (__bf16*)p; p += NL * 2;
  __bf16* w2 = (__bf16*)p; p += NL * 2;
  __bf16* zn = (__bf16*)p; p += NX * 2;   // normalized activations (reused)
  float*  yb = (float*)p;  p += NX * 4;   // q_w projection output
  __bf16* at = (__bf16*)p; p += NX * 2;   // attention output
  float*  x1 = (float*)p;  p += NX * 4;   // x + attn residual
  __bf16* hb = (__bf16*)p; p += NH * 2;   // gelu(lin1) output

  const int cols = BB * CCH * WLEN;  // 8192 norm columns

  // 1. weights -> bf16
  cvt_bf16_kernel<<<dim3((NQ + 255) / 256), 256, 0, stream>>>(q_w, wq, NQ);
  cvt_bf16_kernel<<<dim3((NQ + 255) / 256), 256, 0, stream>>>(out_w, wo, NQ);
  cvt_bf16_kernel<<<dim3((NL + 255) / 256), 256, 0, stream>>>(lin1_w, w1, NL);
  cvt_bf16_kernel<<<dim3((NL + 255) / 256), 256, 0, stream>>>(lin2_w, w2, NL);

  // 2. norm1(x) -> zn
  frame_norm_kernel<<<dim3(cols), 256, 0, stream>>>(x, norm1_w, norm1_b, zn);

  // 3. y = q_w @ zn  (f32 out)
  gemm_bf16_kernel<<<dim3(BINSN / 128, WLEN / 128, BB * CCH), 256, 0, stream>>>(
      wq, zn, nullptr, yb, BINSN, BINSN, WLEN, 1);

  // 4. attention with rotary + online softmax -> at (bf16)
  attn_kernel<<<dim3(WLEN / 128, NHEAD, BB * CCH), 256, 81920, stream>>>(
      yb, rotf, at);

  // 5. x1 = x + out_w @ at
  gemm_bf16_kernel<<<dim3(BINSN / 128, WLEN / 128, BB * CCH), 256, 0, stream>>>(
      wo, at, x, x1, BINSN, BINSN, WLEN, 3);

  // 6. norm2(x1) -> zn
  frame_norm_kernel<<<dim3(cols), 256, 0, stream>>>(x1, norm2_w, norm2_b, zn);

  // 7. h = gelu(lin1 @ zn) -> bf16
  gemm_bf16_kernel<<<dim3(GMLP / 128, WLEN / 128, BB * CCH), 256, 0, stream>>>(
      w1, zn, nullptr, hb, GMLP, BINSN, WLEN, 2);

  // 8. out = x1 + lin2 @ h
  gemm_bf16_kernel<<<dim3(BINSN / 128, WLEN / 128, BB * CCH), 256, 0, stream>>>(
      w2, hb, x1, outp, BINSN, GMLP, WLEN, 3);
}
